// CombinedModel_74096775790955
// MI455X (gfx1250) — matrix operand order repaired and checked by reading, not compiled, not run on hardware
//
#include <hip/hip_runtime.h>
#include <hip/hip_bf16.h>

// Problem constants from the reference
#define IQ 64
#define CH 256
#define TS 1024
#define NT (IQ * TS)   // 65536 total sequential timesteps

typedef __attribute__((ext_vector_type(16))) __bf16          v16bf;
typedef __attribute__((ext_vector_type(16))) unsigned short  v16u;
typedef __attribute__((ext_vector_type(8)))  unsigned short  v8u;
typedef __attribute__((ext_vector_type(8)))  float           v8f;

// float -> bf16, round-to-nearest-even, integer domain.
__device__ __forceinline__ unsigned short f2bf(float f) {
    union { float f; unsigned u; } cv; cv.f = f;
    unsigned u = cv.u;
    u += 0x7FFFu + ((u >> 16) & 1u);
    return (unsigned short)(u >> 16);
}

// ---------------------------------------------------------------------------
// Prep A: convert weights M, H to bf16 (row-major [c, k], same as f32 layout).
// ---------------------------------------------------------------------------
__global__ __launch_bounds__(256) void cvt_weights(const float* __restrict__ M,
                                                   const float* __restrict__ H,
                                                   unsigned short* __restrict__ Mbf,
                                                   unsigned short* __restrict__ Hbf) {
    int i = blockIdx.x * 256 + threadIdx.x;
    Mbf[i] = f2bf(M[i]);
    Hbf[i] = f2bf(H[i]);
}

// ---------------------------------------------------------------------------
// Prep B: u [Iq, C, T] f32  ->  Au [t, k] bf16 (t = iq*T + time, k = channel).
// 32x32 LDS tile transpose: coalesced reads along time, packed writes along k.
// ---------------------------------------------------------------------------
__global__ __launch_bounds__(256) void transpose_u(const float* __restrict__ u,
                                                   unsigned short* __restrict__ Au) {
    __shared__ unsigned short tile[32][33];
    const int t0 = blockIdx.x * 32;          // tile never crosses an iq (T%32==0)
    const int k0 = blockIdx.y * 32;
    const int iq  = t0 >> 10;
    const int tim = t0 & (TS - 1);
    const float* src = u + (long)iq * CH * TS + tim;   // + k*TS + tt

#pragma unroll
    for (int r = 0; r < 4; ++r) {
        int kk = threadIdx.y * 4 + r;
        int tt = threadIdx.x;
        tile[kk][tt] = f2bf(src[(long)(k0 + kk) * TS + tt]);
    }
    __syncthreads();
#pragma unroll
    for (int r = 0; r < 4; ++r) {
        int tt = threadIdx.y * 4 + r;
        int kk = threadIdx.x;
        Au[(long)(t0 + tt) * CH + (k0 + kk)] = tile[kk][tt];
    }
}

// ---------------------------------------------------------------------------
// Fragment loaders (pure 128-bit vector loads, no conversion on the hot path).
// A: lane element e -> k = k0 + e + half*16 == one contiguous 16-element run.
// B: lane element e -> k = k0 + e + half*16 == one contiguous 16-element run.
// ---------------------------------------------------------------------------
__device__ __forceinline__ v16bf load_afrag(const unsigned short* __restrict__ arow,
                                            int k0, int half) {
    v16u a = *(const v16u*)(arow + k0 + half * 16);
    return __builtin_bit_cast(v16bf, a);
}
__device__ __forceinline__ v16bf load_bfrag(const unsigned short* __restrict__ Bw,
                                            int c, int k0, int half) {
    v16u b = *(const v16u*)(Bw + (long)c * CH + k0 + half * 16);
    return __builtin_bit_cast(v16bf, b);
}

// ---------------------------------------------------------------------------
// Kernel: Mu[t, c] = sum_k Au[t,k] * Mbf[c,k].
// Block = 4 waves; each wave owns one 16-row t-tile x 4 c-tiles (A reused 4x),
// K=256 fully unrolled -> 32 v_wmma back-to-back per wave.
// ---------------------------------------------------------------------------
__global__ __launch_bounds__(128) void gemm_mu(const unsigned short* __restrict__ Au,
                                               const unsigned short* __restrict__ Mbf,
                                               float* __restrict__ Mu) {
    const int lane  = threadIdx.x & 31;
    const int wid   = threadIdx.x >> 5;
    const int half  = lane >> 4;
    const int l16   = lane & 15;
    const int tt    = blockIdx.x * 16;
    const int cbase = wid * 64;                       // 4 c-tiles per wave
    const unsigned short* arow = Au + (long)(tt + l16) * CH;

    v8f acc[4] = {};
#pragma unroll
    for (int k0 = 0; k0 < CH; k0 += 32) {
        v16bf a = load_afrag(arow, k0, half);
#pragma unroll
        for (int j = 0; j < 4; ++j) {
            v16bf b = load_bfrag(Mbf, cbase + j * 16 + l16, k0, half);
            acc[j] = __builtin_amdgcn_wmma_f32_16x16x32_bf16(
                false, a, false, b, (short)0, acc[j], false, false);
        }
    }
#pragma unroll
    for (int j = 0; j < 4; ++j)
#pragma unroll
        for (int r = 0; r < 8; ++r)
            Mu[(long)(tt + r + half * 8) * CH + (cbase + j * 16 + l16)] = acc[j][r];
}

// ---------------------------------------------------------------------------
// Sequential recurrence (single workgroup, exact f32).
//   x_{t+1} = x + dt*(-x + W @ tanh(x) + Mu_t)
// 1024 threads; thread (q,c) keeps W[c, q*64..] in VGPRs; x in registers of
// threads 0..255; tanh fused into the update stage -> 2 barriers/step.
// Emits bf16 state Xbf for the readout GEMM + f32 membranes in [Iq,C,T].
// ---------------------------------------------------------------------------
__global__ __launch_bounds__(1024) void scan_rnn(const float* __restrict__ Mu,
                                                 const float* __restrict__ W,
                                                 const float* __restrict__ dtp,
                                                 unsigned short* __restrict__ Xbf,
                                                 float* __restrict__ mem_out) {
    __shared__ float th[CH];        // tanh(x), produced by the update stage
    __shared__ float part[4][CH];   // partial dot products

    const int tid = threadIdx.x;
    const int c   = tid & (CH - 1);
    const int q   = tid >> 8;
    const float dt = dtp[0];

    float wreg[64];
#pragma unroll
    for (int k = 0; k < 64; ++k) wreg[k] = W[c * CH + q * 64 + k];

    float x = 0.0f;
    if (tid < CH) th[c] = 0.0f;     // tanh(0)
    __syncthreads();

    for (int t = 0; t < NT; ++t) {
        float p = 0.0f;
#pragma unroll
        for (int k = 0; k < 64; ++k) p = fmaf(wreg[k], th[q * 64 + k], p);
        part[q][c] = p;
        if (tid < CH && t + 4 < NT)                      // pull Mu ahead into L2/L0
            __builtin_prefetch(&Mu[(long)(t + 4) * CH + c], 0, 0);
        __syncthreads();

        if (tid < CH) {
            float wx = part[0][c] + part[1][c] + part[2][c] + part[3][c];
            x = x + dt * (-x + wx + Mu[(long)t * CH + c]);
            th[c] = tanhf(x);
            Xbf[(long)t * CH + c] = f2bf(x);
            mem_out[(long)(t >> 10) * CH * TS + (long)c * TS + (t & (TS - 1))] = x;
        }
        __syncthreads();
    }
}

// ---------------------------------------------------------------------------
// Readout: Y[t,c] = sigmoid(sum_k Xbf[t,k]*Hbf[c,k] + b[c]) -> [Iq, C, T].
// Same 32-wmma structure as gemm_mu.
// ---------------------------------------------------------------------------
__global__ __launch_bounds__(128) void gemm_readout(const unsigned short* __restrict__ Xbf,
                                                    const unsigned short* __restrict__ Hbf,
                                                    const float* __restrict__ bias,
                                                    float* __restrict__ out) {
    const int lane  = threadIdx.x & 31;
    const int wid   = threadIdx.x >> 5;
    const int half  = lane >> 4;
    const int l16   = lane & 15;
    const int tt    = blockIdx.x * 16;
    const int cbase = wid * 64;
    const unsigned short* arow = Xbf + (long)(tt + l16) * CH;

    v8f acc[4] = {};
#pragma unroll
    for (int k0 = 0; k0 < CH; k0 += 32) {
        v16bf a = load_afrag(arow, k0, half);
#pragma unroll
        for (int j = 0; j < 4; ++j) {
            v16bf b = load_bfrag(Hbf, cbase + j * 16 + l16, k0, half);
            acc[j] = __builtin_amdgcn_wmma_f32_16x16x32_bf16(
                false, a, false, b, (short)0, acc[j], false, false);
        }
    }
#pragma unroll
    for (int j = 0; j < 4; ++j) {
        const int   cc = cbase + j * 16 + l16;
        const float bb = bias[cc];
#pragma unroll
        for (int r = 0; r < 8; ++r) {
            int   t = tt + r + half * 8;
            float z = acc[j][r] + bb;
            float y = 1.0f / (1.0f + __expf(-z));
            out[(long)(t >> 10) * CH * TS + (long)cc * TS + (t & (TS - 1))] = y;
        }
    }
}

// ---------------------------------------------------------------------------
// Inputs (setup_inputs order): u, dt, W, M, H, b.
// Outputs: [outputs | membranes], each [Iq, C, T] f32.
// Workspace: Mu f32 (64MB) + Au bf16 (32MB) + Xbf bf16 (32MB) + Mbf/Hbf (256KB).
// ---------------------------------------------------------------------------
extern "C" void kernel_launch(void* const* d_in, const int* in_sizes, int n_in,
                              void* d_out, int out_size, void* d_ws, size_t ws_size,
                              hipStream_t stream) {
    const float* u  = (const float*)d_in[0];
    const float* dt = (const float*)d_in[1];
    const float* W  = (const float*)d_in[2];
    const float* M  = (const float*)d_in[3];
    const float* H  = (const float*)d_in[4];
    const float* b  = (const float*)d_in[5];

    float* out_y = (float*)d_out;                        // outputs  [Iq,C,T]
    float* out_x = out_y + (size_t)IQ * CH * TS;         // membranes[Iq,C,T]

    const size_t ntch = (size_t)NT * CH;
    float*          Mu  = (float*)d_ws;                        // 64 MB
    unsigned short* Au  = (unsigned short*)(Mu + ntch);        // 32 MB
    unsigned short* Xbf = Au + ntch;                           // 32 MB
    unsigned short* Mbf = Xbf + ntch;                          // 128 KB
    unsigned short* Hbf = Mbf + (size_t)CH * CH;               // 128 KB

    cvt_weights<<<(CH * CH) / 256, 256, 0, stream>>>(M, H, Mbf, Hbf);
    transpose_u<<<dim3(NT / 32, CH / 32), dim3(32, 8), 0, stream>>>(u, Au);
    gemm_mu<<<NT / 16, 128, 0, stream>>>(Au, Mbf, Mu);
    scan_rnn<<<1, 1024, 0, stream>>>(Mu, W, dt, Xbf, out_x);
    gemm_readout<<<NT / 16, 128, 0, stream>>>(Xbf, Hbf, b, out_y);
}
